// PatchSoftUnetGeneratorShiftTriple_41970420416958
// MI455X (gfx1250) — matrix operand order repaired and checked by reading, not compiled, not run on hardware
//
#include <hip/hip_runtime.h>
#include <hip/hip_bf16.h>
#include <math.h>

typedef __bf16 bf16_t;
typedef __attribute__((ext_vector_type(16))) __bf16 v16bf;
typedef __attribute__((ext_vector_type(8)))  float  v8f;

#define HW   4096
#define CH   128
#define HWB  65536   // HW*HW/256 blocks

// ---------------------------------------------------------------------------
// Stage 0a: per-pixel channel sum-of-squares of latter, plus bf16 conversions
//   Fb[p][c] = bf16(former[c][p])   (A of GEMM1, row-major M=p, K=c)
//   Lb[c][q] = bf16(latter[c][q])   (B of GEMM1 and A of GEMM2)
// ---------------------------------------------------------------------------
__global__ __launch_bounds__(256) void conv_prep(
    const float* __restrict__ former, const float* __restrict__ latter,
    bf16_t* __restrict__ Fb, bf16_t* __restrict__ Lb, float* __restrict__ ssum)
{
    int q = blockIdx.x * 256 + threadIdx.x;
    float s = 0.f;
    for (int c = 0; c < CH; ++c) {
        float lv = latter[c * HW + q];
        float fv = former[c * HW + q];
        s += lv * lv;
        Lb[c * HW + q]  = (bf16_t)lv;
        Fb[q * CH + c]  = (bf16_t)fv;
    }
    ssum[q] = s;
}

// Stage 0b: den_inv[q] = 1/max(sqrt(box3x3(ssum)),1e-4); mm[q] from mask mean
__global__ __launch_bounds__(256) void den_mask(
    const float* __restrict__ ssum, const float* __restrict__ msk,
    float* __restrict__ den_inv, float* __restrict__ mmv)
{
    int q = blockIdx.x * 256 + threadIdx.x;
    int y = q >> 6, x = q & 63;
    float s = 0.f, mv = 0.f;
    for (int dy = -1; dy <= 1; ++dy)
        for (int dx = -1; dx <= 1; ++dx) {
            int yy = y + dy, xx = x + dx;
            if ((unsigned)yy < 64u && (unsigned)xx < 64u) {
                int qq = (yy << 6) | xx;
                s  += ssum[qq];
                mv += msk[qq];
            }
        }
    float den = fmaxf(sqrtf(s), 1e-4f);
    den_inv[q] = 1.f / den;
    mmv[q] = (mv * (1.f / 9.f) <= (1.f / 9.f)) ? 1.f : 0.f;
}

// ---------------------------------------------------------------------------
// bf16 WMMA GEMM: C(MxN,f32) = A(MxK) @ B(KxN), row-major, all dims % 64/32.
// 128 threads = 4 waves; block tile 64x64; each wave 32x32 = 2x2 WMMA tiles.
// ---------------------------------------------------------------------------
__global__ __launch_bounds__(128) void gemm_bf16_wmma(
    const bf16_t* __restrict__ A, const bf16_t* __restrict__ B,
    float* __restrict__ C, int K, int ldA, int ldB, int ldC)
{
    __shared__ alignas(16) bf16_t As[64][40];   // [m][k], row stride 80B (16B-aligned)
    __shared__ alignas(16) bf16_t Bs[64][34];   // transposed: [n][k]

    const int tid  = threadIdx.x;
    const int wave = tid >> 5, lane = tid & 31;
    const int l15  = lane & 15;
    const int hi   = (lane & 16) ? 1 : 0;
    const int row0 = blockIdx.y * 64, col0 = blockIdx.x * 64;
    const int wm   = (wave >> 1) * 32, wn = (wave & 1) * 32;

    v8f acc[2][2] = {};
    union U4 { uint4 u4; unsigned u[4]; bf16_t h[8]; };
    union FR { v16bf v; unsigned u[8]; };

    for (int k0 = 0; k0 < K; k0 += 32) {
        // ---- stage A 64x32 via b128 loads
        for (int i = tid; i < 256; i += 128) {
            int r = i >> 2, c8 = (i & 3) << 3;
            U4 v; v.u4 = *(const uint4*)(A + (size_t)(row0 + r) * ldA + k0 + c8);
            *(uint4*)(&As[r][c8]) = v.u4;
        }
        // ---- stage B 32x64 via b128 loads, transpose into Bs[n][k]
        for (int i = tid; i < 256; i += 128) {
            int kk = i >> 3, n8 = (i & 7) << 3;
            U4 v; v.u4 = *(const uint4*)(B + (size_t)(k0 + kk) * ldB + col0 + n8);
            #pragma unroll
            for (int e = 0; e < 8; ++e) Bs[n8 + e][kk] = v.h[e];
        }
        if (k0 + 32 < K) {  // prefetch next K tile (global_prefetch_b8)
            __builtin_prefetch(A + (size_t)(row0 + (tid & 63)) * ldA + k0 + 32, 0, 1);
            __builtin_prefetch(B + (size_t)(k0 + 32 + (tid & 31)) * ldB + col0, 0, 1);
        }
        __syncthreads();

        // ---- per-lane fragments per documented 16-bit A/B layouts
        FR af[2], bfr[2];
        const int kbA = hi ? 8 : 0;
        #pragma unroll
        for (int r = 0; r < 2; ++r) {
            int m = wm + 16 * r + l15;
            #pragma unroll
            for (int j = 0; j < 8; ++j) {
                int kk = kbA + 2 * j + ((j >= 4) ? 8 : 0);   // K pairs 0..7 / 16..23 (+8 hi)
                af[r].u[j] = *(const unsigned*)(&As[m][kk]);
            }
        }
        const int kbB = hi ? 8 : 0;
        #pragma unroll
        for (int c = 0; c < 2; ++c) {
            int n = wn + 16 * c + l15;
            #pragma unroll
            for (int j = 0; j < 8; ++j)
                bfr[c].u[j] = *(const unsigned*)(&Bs[n][kbB + 2 * j + ((j >= 4) ? 8 : 0)]);
        }
        #pragma unroll
        for (int r = 0; r < 2; ++r)
            #pragma unroll
            for (int c = 0; c < 2; ++c)
                acc[r][c] = __builtin_amdgcn_wmma_f32_16x16x32_bf16(
                    false, af[r].v, false, bfr[c].v, (short)0, acc[r][c], false, false);
        __syncthreads();
    }

    // ---- epilogue: D layout VGPR v -> M = v + 8*hi, N = lane&15
    #pragma unroll
    for (int r = 0; r < 2; ++r)
        #pragma unroll
        for (int c = 0; c < 2; ++c) {
            int n = col0 + wn + 16 * c + l15;
            #pragma unroll
            for (int v = 0; v < 8; ++v) {
                int m = row0 + wm + 16 * r + v + hi * 8;
                C[(size_t)m * ldC + n] = acc[r][c][v];
            }
        }
}

// ---------------------------------------------------------------------------
// scores[p][w] = den_inv[w] * sum_{dy,dx in [-1,1]} G[p+d][w+d] (2D-valid both)
// ---------------------------------------------------------------------------
__global__ __launch_bounds__(256) void shift9_scale(
    const float* __restrict__ G, float* __restrict__ S, const float* __restrict__ den_inv)
{
    int idx = blockIdx.x * 256 + threadIdx.x;
    int p = idx >> 12, w = idx & 4095;
    int yp = p >> 6, xp = p & 63, yw = w >> 6, xw = w & 63;
    float s = 0.f;
    #pragma unroll
    for (int dy = -1; dy <= 1; ++dy)
        #pragma unroll
        for (int dx = -1; dx <= 1; ++dx) {
            int a = yp + dy, bx = xp + dx, cy = yw + dy, d = xw + dx;
            if ((unsigned)a < 64u && (unsigned)bx < 64u && (unsigned)cy < 64u && (unsigned)d < 64u)
                s += G[(size_t)((a << 6) | bx) * HW + ((cy << 6) | d)];
        }
    S[idx] = s * den_inv[w];
}

// First flat diag conv (reference's transpose folded in): U1[p][w]=sum_d S[p+d][w+d]
__global__ __launch_bounds__(256) void flatdiag1(
    const float* __restrict__ In, float* __restrict__ Out)
{
    int idx = blockIdx.x * 256 + threadIdx.x;
    int p = idx >> 12, w = idx & 4095;
    float s = In[idx];
    if (p >= 1 && w >= 1)       s += In[idx - (HW + 1)];
    if (p < HW - 1 && w < HW - 1) s += In[idx + (HW + 1)];
    Out[idx] = s;
}

// Second flat diag conv in x-major flat index, result back in (p,w) orientation
__global__ __launch_bounds__(256) void flatdiag2(
    const float* __restrict__ In, float* __restrict__ Out)
{
    int idx = blockIdx.x * 256 + threadIdx.x;
    int p = idx >> 12, w = idx & 4095;
    int fp = ((p & 63) << 6) | (p >> 6);   // x-major flat index
    int fw = ((w & 63) << 6) | (w >> 6);
    float s = 0.f;
    #pragma unroll
    for (int d = -1; d <= 1; ++d) {
        int fpd = fp + d, fwd = fw + d;
        if ((unsigned)fpd < (unsigned)HW && (unsigned)fwd < (unsigned)HW) {
            int pp = ((fpd & 63) << 6) | (fpd >> 6);
            int ww = ((fwd & 63) << 6) | (fwd >> 6);
            s += In[(size_t)pp * HW + ww];
        }
    }
    Out[idx] = s;
}

// Row softmax over w (the reference's axis-0 softmax, stored transposed):
// cos[p][w] = softmax_w(V[p][w]*mm[w]*10) * mm[w]
__global__ __launch_bounds__(256) void softmax_rows(
    const float* __restrict__ V, float* __restrict__ Cos, const float* __restrict__ mmv)
{
    __shared__ float red[8];
    const int p = blockIdx.x, tid = threadIdx.x;
    const float* row = V + (size_t)p * HW;
    float lmax = -3.402823e38f;
    for (int w = tid; w < HW; w += 256)
        lmax = fmaxf(lmax, row[w] * mmv[w] * 10.f);
    for (int o = 16; o > 0; o >>= 1) lmax = fmaxf(lmax, __shfl_down(lmax, o, 32));
    if ((tid & 31) == 0) red[tid >> 5] = lmax;
    __syncthreads();
    float m = red[0];
    for (int i = 1; i < 8; ++i) m = fmaxf(m, red[i]);
    __syncthreads();
    float lsum = 0.f;
    for (int w = tid; w < HW; w += 256)
        lsum += __expf(row[w] * mmv[w] * 10.f - m);
    for (int o = 16; o > 0; o >>= 1) lsum += __shfl_down(lsum, o, 32);
    if ((tid & 31) == 0) red[tid >> 5] = lsum;
    __syncthreads();
    float tot = 0.f;
    for (int i = 0; i < 8; ++i) tot += red[i];
    float inv = 1.f / tot;
    for (int w = tid; w < HW; w += 256)
        Cos[(size_t)p * HW + w] = __expf(row[w] * mmv[w] * 10.f - m) * inv * mmv[w];
}

// M[k][q] = (1/9) sum_d cos[k-d][q-d]  (cos stored as [p][w] -> read [q-d][k-d]), bf16
__global__ __launch_bounds__(256) void build_M(
    const float* __restrict__ CosPW, bf16_t* __restrict__ Mb)
{
    int idx = blockIdx.x * 256 + threadIdx.x;
    int k = idx >> 12, q = idx & 4095;
    int yk = k >> 6, xk = k & 63, yq = q >> 6, xq = q & 63;
    float s = 0.f;
    #pragma unroll
    for (int dy = -1; dy <= 1; ++dy)
        #pragma unroll
        for (int dx = -1; dx <= 1; ++dx) {
            int a = yk - dy, bx = xk - dx, cy = yq - dy, d = xq - dx;
            if ((unsigned)a < 64u && (unsigned)bx < 64u && (unsigned)cy < 64u && (unsigned)d < 64u)
                s += CosPW[(size_t)((cy << 6) | d) * HW + ((a << 6) | bx)];
        }
    Mb[idx] = (bf16_t)(s * (1.f / 9.f));
}

// ---------------------------------------------------------------------------
extern "C" void kernel_launch(void* const* d_in, const int* in_sizes, int n_in,
                              void* d_out, int out_size, void* d_ws, size_t ws_size,
                              hipStream_t stream)
{
    const float* x    = (const float*)d_in[0];   // (4,256,64,64)
    const float* mask = (const float*)d_in[1];   // (4,1,64,64)
    float* out        = (float*)d_out;           // (4,384,64,64)
    (void)in_sizes; (void)n_in; (void)out_size; (void)ws_size;

    size_t off = 0;
    auto carve = [&](size_t bytes) {
        void* p = (char*)d_ws + off;
        off += (bytes + 255) & ~(size_t)255;
        return p;
    };
    float*  B0      = (float*)carve((size_t)HW * HW * 4);   // 64 MB
    float*  B1      = (float*)carve((size_t)HW * HW * 4);   // 64 MB
    bf16_t* Mb      = (bf16_t*)carve((size_t)HW * HW * 2);  // 32 MB
    bf16_t* Fb      = (bf16_t*)carve((size_t)HW * CH * 2);
    bf16_t* Lb      = (bf16_t*)carve((size_t)CH * HW * 2);
    float*  ssum    = (float*)carve(HW * 4);
    float*  den_inv = (float*)carve(HW * 4);
    float*  mmv     = (float*)carve(HW * 4);

    for (int s = 0; s < 4; ++s) {
        const float* former = x + (size_t)s * 256 * HW;
        const float* latter = former + (size_t)CH * HW;
        const float* msk    = mask + (size_t)s * HW;
        float* outS         = out + (size_t)s * 384 * HW;

        // pass-through channels 0..255
        hipMemcpyAsync(outS, former, (size_t)256 * HW * sizeof(float),
                       hipMemcpyDeviceToDevice, stream);

        conv_prep<<<HW / 256, 256, 0, stream>>>(former, latter, Fb, Lb, ssum);
        den_mask <<<HW / 256, 256, 0, stream>>>(ssum, msk, den_inv, mmv);

        // G = former^T @ latter  (4096x4096, K=128)
        gemm_bf16_wmma<<<dim3(HW / 64, HW / 64), 128, 0, stream>>>(
            Fb, Lb, B0, CH, CH, HW, HW);

        shift9_scale<<<HWB, 256, 0, stream>>>(B0, B1, den_inv);
        flatdiag1   <<<HWB, 256, 0, stream>>>(B1, B0);
        flatdiag2   <<<HWB, 256, 0, stream>>>(B0, B1);
        softmax_rows<<<HW, 256, 0, stream>>>(B1, B0, mmv);
        build_M     <<<HWB, 256, 0, stream>>>(B0, Mb);

        // shift = latter @ M  (128x4096, K=4096) -> channels 256..383
        gemm_bf16_wmma<<<dim3(HW / 64, CH / 64), 128, 0, stream>>>(
            Lb, Mb, outS + (size_t)256 * HW, HW, HW, HW, HW);
    }
}
